// MoEFFNLayer_88338887344224
// MI455X (gfx1250) — matrix operand order repaired and checked by reading, not compiled, not run on hardware
//
#include <hip/hip_runtime.h>
#include <hip/hip_bf16.h>
#include <math.h>

#define T_TOKENS 2048
#define HIDDEN   1024
#define D_FF     2816
#define N_EXP    8
#define TOPK     2
#define NCHUNK   (D_FF / 128)

typedef __attribute__((ext_vector_type(16))) __bf16 v16bf;
typedef __attribute__((ext_vector_type(8)))  float  v8f;

union ABu { v16bf v; unsigned int u[8]; };

__device__ __forceinline__ unsigned int f32_to_bf16_rne(float f) {
  union { float f; unsigned int u; } v; v.f = f;
  unsigned int lsb = (v.u >> 16) & 1u;
  return (v.u + 0x7fffu + lsb) >> 16;
}
__device__ __forceinline__ unsigned int pack2_bf16(float lo, float hi) {
  return f32_to_bf16_rne(lo) | (f32_to_bf16_rne(hi) << 16);
}

__device__ __forceinline__ v8f wmma_bf16(v16bf a, v16bf b, v8f c) {
  // 8 args: (neg_a, A, neg_b, B, c_mod, C, reuse_a, reuse_b)
  return __builtin_amdgcn_wmma_f32_16x16x32_bf16(false, a, false, b, (short)0, c,
                                                 false, false);
}

// A operand (16x32 bf16) from LDS, per ISA 16-bit A layout:
// lanes 0-15 (M=lane): V0..3 = K0..7, V4..7 = K16..23
// lanes 16-31 (M=lane-16): V0..3 = K8..15, V4..7 = K24..31
__device__ __forceinline__ v16bf load_A_lds(const unsigned short* rowbase, int lane) {
  const unsigned int* p = (const unsigned int*)rowbase;  // row at k0 (4B aligned)
  const int hs = (lane >> 4) << 2;
  ABu a;
#pragma unroll
  for (int j = 0; j < 4; ++j) a.u[j] = p[hs + j];
#pragma unroll
  for (int j = 0; j < 4; ++j) a.u[4 + j] = p[8 + hs + j];
  return a.v;
}

// B operand (32x16): lane%16 = N col, elements follow A's K order (row = k0 + lane).
__device__ __forceinline__ v16bf load_B_bf16(const unsigned short* base, int stride,
                                             int row, int n0) {
  const int kb = (row & ~31) + ((row >> 4) & 1) * 8;
  const unsigned short* q = base + (size_t)kb * stride + n0 + (row & 15);
  ABu b;
#pragma unroll
  for (int j = 0; j < 8; ++j) {
    const int ko = 2 * j + ((2 * j >= 8) ? 8 : 0);
    b.u[j] = (unsigned int)q[(size_t)ko * stride] |
             ((unsigned int)q[(size_t)(ko + 1) * stride] << 16);
  }
  return b.v;
}
__device__ __forceinline__ v16bf load_B_f32(const float* base, int stride, int row,
                                            int n0) {
  const int kb = (row & ~31) + ((row >> 4) & 1) * 8;
  const float* q = base + (size_t)kb * stride + n0 + (row & 15);
  ABu b;
#pragma unroll
  for (int j = 0; j < 8; ++j) {
    const int ko = 2 * j + ((2 * j >= 8) ? 8 : 0);
    b.u[j] = pack2_bf16(q[(size_t)ko * stride], q[(size_t)(ko + 1) * stride]);
  }
  return b.v;
}

// ---------------- init: zero output + routing state ----------------
__global__ void init_kernel(float* out, int n, int* counts, float* importance) {
  int i = blockIdx.x * blockDim.x + threadIdx.x;
  if (i < n) out[i] = 0.f;
  if (i < N_EXP) { counts[i] = 0; importance[i] = 0.f; }
}

// ---------------- gating: one wave32 per token ----------------
__global__ __launch_bounds__(256) void gating_kernel(
    const float* __restrict__ x, const float* __restrict__ gw,
    int* __restrict__ counts, int* __restrict__ tokidx, float* __restrict__ wts,
    float* __restrict__ importance) {
  const int t = blockIdx.x * 8 + (threadIdx.x >> 5);
  const int lane = threadIdx.x & 31;
  const float* xr = x + (size_t)t * HIDDEN;
  float acc[N_EXP];
#pragma unroll
  for (int e = 0; e < N_EXP; ++e) acc[e] = 0.f;
  for (int h = lane; h < HIDDEN; h += 32) {
    float xv = xr[h];
#pragma unroll
    for (int e = 0; e < N_EXP; ++e) acc[e] += xv * gw[e * HIDDEN + h];
  }
#pragma unroll
  for (int e = 0; e < N_EXP; ++e) {
#pragma unroll
    for (int m = 16; m >= 1; m >>= 1) acc[e] += __shfl_xor(acc[e], m, 32);
  }
  if (lane == 0) {
    int i0 = 0;
#pragma unroll
    for (int e = 1; e < N_EXP; ++e)
      if (acc[e] > acc[i0]) i0 = e;
    int i1 = (i0 == 0) ? 1 : 0;
#pragma unroll
    for (int e = 0; e < N_EXP; ++e)
      if (e != i0 && acc[e] > acc[i1]) i1 = e;
    float p0 = 1.f / (1.f + __expf(acc[i1] - acc[i0]));
    float p1 = 1.f - p0;
    int s0 = atomicAdd(&counts[i0], 1);
    tokidx[i0 * T_TOKENS + s0] = t; wts[i0 * T_TOKENS + s0] = p0;
    int s1 = atomicAdd(&counts[i1], 1);
    tokidx[i1 * T_TOKENS + s1] = t; wts[i1 * T_TOKENS + s1] = p1;
    // full softmax for importance (aux loss)
    float mx = acc[0];
#pragma unroll
    for (int e = 1; e < N_EXP; ++e) mx = fmaxf(mx, acc[e]);
    float p[N_EXP], s = 0.f;
#pragma unroll
    for (int e = 0; e < N_EXP; ++e) { p[e] = __expf(acc[e] - mx); s += p[e]; }
    float inv = 1.f / s;
#pragma unroll
    for (int e = 0; e < N_EXP; ++e) atomicAdd(&importance[e], p[e] * inv);
  }
}

__global__ void finalize_kernel(const int* counts, const float* importance,
                                float* out_aux) {
  if (blockIdx.x == 0 && threadIdx.x == 0) {
    float aux = 0.f;
    for (int e = 0; e < N_EXP; ++e) {
      float usage = (float)counts[e] / ((float)(T_TOKENS * TOPK) + 1e-9f);
      float imp = importance[e] / (float)T_TOKENS;
      aux += usage * imp;
    }
    aux *= (float)N_EXP;
    out_aux[0] = fminf(aux * 0.01f, 1.f);
  }
}

// ---------------- weight fp32 -> bf16 pre-conversion ----------------
__global__ void cvt_kernel(const float* __restrict__ src,
                           unsigned short* __restrict__ dst, long long n) {
  long long i = (long long)blockIdx.x * blockDim.x + threadIdx.x;
  if (i < (n >> 1))
    ((unsigned int*)dst)[i] = pack2_bf16(src[2 * i], src[2 * i + 1]);
}

// ---------------- fused expert FFN (grouped GEMM, WMMA bf16) ----------------
// CTA: 256 threads = 8 waves. M = 64 gathered tokens. Wave w owns output cols
// [w*128, w*128+128). F processed in 128-wide chunks: phase A builds bf16 act in
// LDS, phase B accumulates act @ w2 in 256 f32 VGPRs per wave. B loads are
// double-buffered (ping-pong, unroll-2 so no register rotation copies); the
// next F-chunk's weights are prefetched via gfx1250 global_prefetch.
template <bool PRE>
__global__ __launch_bounds__(256) void moe_ffn_kernel(
    const float* __restrict__ x, const float* __restrict__ w1,
    const float* __restrict__ w3, const float* __restrict__ w2,
    const unsigned short* __restrict__ w1b, const unsigned short* __restrict__ w3b,
    const unsigned short* __restrict__ w2b, const int* __restrict__ counts,
    const int* __restrict__ tok, const float* __restrict__ wts,
    float* __restrict__ out) {
  const int e = blockIdx.y;
  const int cnt = counts[e];
  const int base = blockIdx.x * 64;
  if (base >= cnt) return;  // uniform across block

  const int tid = threadIdx.x;
  const int wv = tid >> 5;
  const int lane = tid & 31;
  const int m16 = lane & 15;

  extern __shared__ char smem[];
  const int XS = HIDDEN + 8;  // padded ushort stride (bank-conflict free A loads)
  const int AS = 128 + 8;
  unsigned short* xs = (unsigned short*)smem;       // [64][XS]
  unsigned short* acts = xs + 64 * XS;              // [64][AS]
  int* tok_s = (int*)(acts + 64 * AS);              // [64]
  float* wt_s = (float*)(tok_s + 64);               // [64]

  if (tid < 64) {
    int slot = base + tid;
    if (slot < cnt) {
      tok_s[tid] = tok[(size_t)e * T_TOKENS + slot];
      wt_s[tid] = wts[(size_t)e * T_TOKENS + slot];
    } else {
      tok_s[tid] = -1;
      wt_s[tid] = 0.f;
    }
  }
  __syncthreads();
  // gather + convert x tile to bf16 in LDS
  for (int i = tid; i < 64 * HIDDEN; i += 256) {
    int row = i >> 10, col = i & (HIDDEN - 1);
    int t = tok_s[row];
    float v = (t >= 0) ? x[(size_t)t * HIDDEN + col] : 0.f;
    xs[row * XS + col] = (unsigned short)f32_to_bf16_rne(v);
  }
  __syncthreads();

  const size_t w1off = (size_t)e * HIDDEN * D_FF;
  const size_t w2off = (size_t)e * D_FF * HIDDEN;
  const float* w1e = w1 + w1off;
  const float* w3e = w3 + w1off;
  const float* w2e = w2 + w2off;
  const unsigned short* w1be = w1b + w1off;
  const unsigned short* w3be = w3b + w1off;
  const unsigned short* w2be = w2b + w2off;

  v8f acc[4][8];
  {
    v8f z = {0.f, 0.f, 0.f, 0.f, 0.f, 0.f, 0.f, 0.f};
#pragma unroll
    for (int mt = 0; mt < 4; ++mt)
#pragma unroll
      for (int nt = 0; nt < 8; ++nt) acc[mt][nt] = z;
  }

  const int rbase = (lane >> 4) * 8;

#pragma unroll 1
  for (int ch = 0; ch < NCHUNK; ++ch) {
    const int f0 = ch * 128;
    const bool pf_ok = (ch + 1) < NCHUNK;  // uniform
    // ---- phase A: act[64][128] = silu(x@w1) * (x@w3), wave w -> cols w*16..+16
    v8f h1a[4], h3a[4];
    {
      v8f z = {0.f, 0.f, 0.f, 0.f, 0.f, 0.f, 0.f, 0.f};
#pragma unroll
      for (int mt = 0; mt < 4; ++mt) { h1a[mt] = z; h3a[mt] = z; }
    }
    const int fcol = f0 + wv * 16;
    // explicit ping-pong double buffer; unroll-2 folds cur/nxt to constants
    v16bf b1[2], b3[2];
    if (PRE) {
      b1[0] = load_B_bf16(w1be, D_FF, lane, fcol);
      b3[0] = load_B_bf16(w3be, D_FF, lane, fcol);
    } else {
      b1[0] = load_B_f32(w1e, D_FF, lane, fcol);
      b3[0] = load_B_f32(w3e, D_FF, lane, fcol);
    }
#pragma unroll 2
    for (int ks = 0; ks < HIDDEN / 32; ++ks) {
      const int cur = ks & 1, nxt = cur ^ 1;
      if (ks + 1 < HIDDEN / 32) {  // issue next K-step's B loads first
        const int nrow = (ks + 1) * 32 + lane;
        if (PRE) {
          b1[nxt] = load_B_bf16(w1be, D_FF, nrow, fcol);
          b3[nxt] = load_B_bf16(w3be, D_FF, nrow, fcol);
        } else {
          b1[nxt] = load_B_f32(w1e, D_FF, nrow, fcol);
          b3[nxt] = load_B_f32(w3e, D_FF, nrow, fcol);
        }
      }
      if (pf_ok && ks < 30) {  // warm GL2 for next F-chunk's weight rows
        const int prow = ks * 32 + lane;
        if (PRE) {
          __builtin_prefetch(w1be + (size_t)prow * D_FF + fcol + 128, 0, 3);
          __builtin_prefetch(w3be + (size_t)prow * D_FF + fcol + 128, 0, 3);
        } else {
          __builtin_prefetch(w1e + (size_t)prow * D_FF + fcol + 128, 0, 3);
          __builtin_prefetch(w3e + (size_t)prow * D_FF + fcol + 128, 0, 3);
        }
      }
#pragma unroll
      for (int mt = 0; mt < 4; ++mt) {
        v16bf a = load_A_lds(xs + (size_t)(mt * 16 + m16) * XS + ks * 32, lane);
        h1a[mt] = wmma_bf16(a, b1[cur], h1a[mt]);
        h3a[mt] = wmma_bf16(a, b3[cur], h3a[mt]);
      }
    }
    // store act (C layout: VGPR r -> row rbase+r, lane%16 -> col)
    const int ccol = wv * 16 + m16;
#pragma unroll
    for (int mt = 0; mt < 4; ++mt) {
#pragma unroll
      for (int r = 0; r < 8; ++r) {
        float g = h1a[mt][r], u = h3a[mt][r];
        // fast sigmoid: v_rcp_f32 instead of IEEE division chain
        float sg = g * __builtin_amdgcn_rcpf(1.f + __expf(-g));
        acts[(size_t)(mt * 16 + rbase + r) * AS + ccol] =
            (unsigned short)f32_to_bf16_rne(sg * u);
      }
    }
    __syncthreads();
    // ---- phase B: acc += act @ w2[f0..f0+128][wv*128..+128]
#pragma unroll 1
    for (int ks = 0; ks < 4; ++ks) {
      v16bf aM[4];
#pragma unroll
      for (int mt = 0; mt < 4; ++mt)
        aM[mt] = load_A_lds(acts + (size_t)(mt * 16 + m16) * AS + ks * 32, lane);
      const int frow = f0 + ks * 32 + lane;
      if (pf_ok) {  // next-chunk w2 rows for this wave's 128 output columns
        if (PRE) {
          const unsigned short* pf =
              w2be + (size_t)(frow + 128) * HIDDEN + wv * 128;
          __builtin_prefetch(pf, 0, 3);
          __builtin_prefetch(pf + 64, 0, 3);
        } else {
          const float* pf = w2e + (size_t)(frow + 128) * HIDDEN + wv * 128;
          __builtin_prefetch(pf, 0, 3);
          __builtin_prefetch(pf + 32, 0, 3);
          __builtin_prefetch(pf + 64, 0, 3);
          __builtin_prefetch(pf + 96, 0, 3);
        }
      }
#pragma unroll
      for (int nt = 0; nt < 8; ++nt) {
        const int n0 = wv * 128 + nt * 16;
        v16bf b = PRE ? load_B_bf16(w2be, HIDDEN, frow, n0)
                      : load_B_f32(w2e, HIDDEN, frow, n0);
#pragma unroll
        for (int mt = 0; mt < 4; ++mt) acc[mt][nt] = wmma_bf16(aM[mt], b, acc[mt][nt]);
      }
    }
    __syncthreads();  // protect acts before next chunk overwrites
  }

  // ---- scatter-add: out[tok] += weight * y
#pragma unroll
  for (int mt = 0; mt < 4; ++mt) {
#pragma unroll
    for (int r = 0; r < 8; ++r) {
      int row = mt * 16 + rbase + r;
      int t = tok_s[row];
      if (t < 0) continue;
      float wgt = wt_s[row];
      float* orow = out + (size_t)t * HIDDEN + m16;
#pragma unroll
      for (int nt = 0; nt < 8; ++nt)
        atomicAdd(orow + wv * 128 + nt * 16, wgt * acc[mt][nt][r]);
    }
  }
}

extern "C" void kernel_launch(void* const* d_in, const int* in_sizes, int n_in,
                              void* d_out, int out_size, void* d_ws, size_t ws_size,
                              hipStream_t stream) {
  const float* x = (const float*)d_in[0];
  const float* gw = (const float*)d_in[1];
  const float* w1 = (const float*)d_in[2];
  const float* w3 = (const float*)d_in[3];
  const float* w2 = (const float*)d_in[4];
  float* out = (float*)d_out;

  char* ws = (char*)d_ws;
  int* counts = (int*)ws;                                   // 32 B
  float* importance = (float*)(ws + 64);                    // 32 B
  int* tokidx = (int*)(ws + 128);                           // E*T ints
  float* wts = (float*)(ws + 128 + (size_t)N_EXP * T_TOKENS * 4);
  size_t woff = 128 + 2ull * N_EXP * T_TOKENS * 4;
  woff = (woff + 511) & ~(size_t)511;
  const size_t welems = (size_t)N_EXP * HIDDEN * D_FF;      // == E*F*H too
  unsigned short* w1b = (unsigned short*)(ws + woff);
  unsigned short* w3b = w1b + welems;
  unsigned short* w2b = w3b + welems;
  const bool pre = ws_size >= woff + 3ull * welems * 2;     // constant per session

  const int n_out = T_TOKENS * HIDDEN + 1;
  init_kernel<<<(n_out + 255) / 256, 256, 0, stream>>>(out, n_out, counts, importance);
  gating_kernel<<<T_TOKENS / 8, 256, 0, stream>>>(x, gw, counts, tokidx, wts,
                                                  importance);
  finalize_kernel<<<1, 32, 0, stream>>>(counts, importance, out + T_TOKENS * HIDDEN);

  if (pre) {
    const int blks = (int)((welems / 2 + 255) / 256);
    cvt_kernel<<<blks, 256, 0, stream>>>(w1, w1b, (long long)welems);
    cvt_kernel<<<blks, 256, 0, stream>>>(w3, w3b, (long long)welems);
    cvt_kernel<<<blks, 256, 0, stream>>>(w2, w2b, (long long)welems);
  }

  dim3 grid(T_TOKENS / 64, N_EXP);
  size_t smem = (size_t)64 * (HIDDEN + 8) * 2 + (size_t)64 * (128 + 8) * 2 + 64 * 4 +
                64 * 4;
  if (pre)
    moe_ffn_kernel<true><<<grid, 256, smem, stream>>>(x, w1, w3, w2, w1b, w3b, w2b,
                                                      counts, tokidx, wts, out);
  else
    moe_ffn_kernel<false><<<grid, 256, smem, stream>>>(x, w1, w3, w2, w1b, w3b, w2b,
                                                       counts, tokidx, wts, out);
}
